// Attention_76089640616322
// MI455X (gfx1250) — matrix operand order repaired and checked by reading, not compiled, not run on hardware
//
#include <hip/hip_runtime.h>
#include <hip/hip_bf16.h>

#define B_   32
#define S_   4096
#define HE_  512
#define HD_  1024

typedef __attribute__((ext_vector_type(16))) __bf16 v16bf;
typedef __attribute__((ext_vector_type(8)))  float  v8f;
typedef __attribute__((ext_vector_type(4)))  float  fx4;

__device__ __forceinline__ unsigned int f2bf(float f) {
  unsigned int u = __float_as_uint(f);
  return (u + 0x7FFFu + ((u >> 16) & 1u)) >> 16;   // round-to-nearest-even bf16
}
__device__ __forceinline__ unsigned int pack2bf(float lo, float hi) {
  return f2bf(lo) | (f2bf(hi) << 16);
}

// branch-free tanh: 1 - 2/(exp(2x)+1) with hardware rcp (v_rcp_f32).
// Correct limits: x->+inf => exp=inf => rcp=0 => 1; x->-inf => exp=0 => -1.
__device__ __forceinline__ float fast_tanh(float x) {
  float e = __expf(2.0f * x);
  return 1.0f - 2.0f * __builtin_amdgcn_rcpf(e + 1.0f);
}

// ---------------------------------------------------------------------------
// Kernel 1: pack W1[:, :HE] (f32 row-major [e, HD]) into the exact B-fragment
// layout for v_wmma_f32_16x16x32_bf16:
//   dword ((et*16 + kk)*32 + lane)*8 + v = {W1[et*16+(lane&15)][d], W1[et*16+(lane&15)][d+1]},
//   d = kk*32 + 8*(lane>>4) + 2v + 8*(v>>2).  Per-lane fragment load = two contiguous b128s.
// ---------------------------------------------------------------------------
__global__ void pack_w1_kernel(const float* __restrict__ W1,
                               unsigned int* __restrict__ W1b) {
  int idx = blockIdx.x * blockDim.x + threadIdx.x;   // 131072 dwords total
  int v  = idx & 7;
  int l  = (idx >> 3) & 31;
  int kk = (idx >> 8) & 15;
  int et = idx >> 12;
  int d  = kk * 32 + 8 * (l >> 4) + 2 * v + ((v >> 2) << 3);
  int e0 = et * 16 + (l & 15);
  float a = W1[(size_t)e0 * HD_ + d];
  float b = W1[(size_t)e0 * HD_ + d + 1];
  W1b[idx] = pack2bf(a, b);
}

// ---------------------------------------------------------------------------
// Kernel 2: per-batch bias from the broadcast half:
//   bias[b][e] = sum_d output[b][d] * W1[e][HE + d]
// ---------------------------------------------------------------------------
__global__ void bias_kernel(const float* __restrict__ outp,
                            const float* __restrict__ W1,
                            float* __restrict__ bias) {
  __shared__ float o[512];
  const int b = blockIdx.x, t = threadIdx.x;   // 256 threads
  o[t]       = outp[b * 512 + t];
  o[t + 256] = outp[b * 512 + t + 256];
  __syncthreads();
  for (int e = t; e < HE_; e += 256) {
    const float* w = W1 + (size_t)e * HD_ + HE_;
    float acc = 0.f;
    for (int d = 0; d < 512; ++d) acc += o[d] * w[d];
    bias[b * HE_ + e] = acc;
  }
}

// ---------------------------------------------------------------------------
// Kernel 3 (hot): fused  score[b,s] = sum_e tanh(ctx@W1a^T + bias[e])*W2[e]
// via v_wmma_f32_16x16x32_bf16.  Block = 256 threads = 8 waves; each wave owns
// a 16-row s-tile staged as bf16 in dynamic LDS.  All 8 waves issue identical
// B-fragment addresses -> W1b streamed once per block through WGP$.
// ---------------------------------------------------------------------------
#define SCORE_SMEM (4096 + 8 * 16 * 520 * 2)   // bias+W2 (4KB) + 8 slabs (130KB)

__global__ __launch_bounds__(256)
void score_kernel(const float* __restrict__ ctx_g,
                  const unsigned int* __restrict__ W1b,
                  const float* __restrict__ bias,
                  const float* __restrict__ W2,
                  float* __restrict__ scores) {
  extern __shared__ char smem[];
  float* biasS = (float*)smem;                       // [512]
  float* w2S   = biasS + HE_;                        // [512]
  unsigned short* ctxs = (unsigned short*)(smem + 4096);   // 8 x 16 x 520

  const int t    = threadIdx.x;
  const int lane = t & 31;
  const int wv   = t >> 5;                 // wave 0..7
  const int b     = blockIdx.x >> 5;       // / (S/128)
  const int stile = blockIdx.x & 31;
  const int s0w   = stile * 128 + wv * 16; // this wave's 16 rows

  for (int e = t; e < HE_; e += 256) { biasS[e] = bias[b * HE_ + e]; w2S[e] = W2[e]; }

  // stage 16 context rows (f32 -> bf16) into LDS, row-major stride 520.
  // Non-temporal: context is a read-once stream; keep WGP$/L2 for W1b.
  {
    const size_t rowbase = ((size_t)b * S_ + s0w) * HE_;
    unsigned short* dst = ctxs + wv * 16 * 520;
    for (int i = lane; i < 16 * 128; i += 32) {
      int r  = i >> 7;
      int c4 = i & 127;
      const fx4 f = __builtin_nontemporal_load(
          (const fx4*)(ctx_g + rowbase + (size_t)r * HE_ + c4 * 4));
      *(uint2*)(dst + r * 520 + c4 * 4) =
          make_uint2(pack2bf(f.x, f.y), pack2bf(f.z, f.w));
    }
  }
  __syncthreads();

  union Frag { uint4 u[2]; v16bf v; };

  const unsigned short* my = ctxs + wv * 16 * 520;
  const int row = lane & 15;       // A layout: lane (mod 16) = matrix row M
  const int hi  = lane >> 4;       // K-half swizzle select
  float sc[8];
  #pragma unroll
  for (int g = 0; g < 8; ++g) sc[g] = 0.f;

  for (int et = 0; et < 32; ++et) {            // 32 e-tiles of 16
    v8f acc = {};
    #pragma unroll 4
    for (int kk = 0; kk < 16; ++kk) {          // K = 512 in steps of 32
      Frag a, bm;
      // A fragment: lane<16 -> K {0..7, 16..23}; lane>=16 -> K {8..15, 24..31}
      const unsigned short* ap = my + row * 520 + kk * 32 + hi * 8;
      a.u[0] = *(const uint4*)(ap);
      a.u[1] = *(const uint4*)(ap + 16);
      // B fragment: pre-packed, two contiguous b128s per lane (same addr all waves)
      const uint4* bp = (const uint4*)(W1b + ((size_t)((et * 16 + kk) * 32 + lane)) * 8);
      bm.u[0] = bp[0];
      bm.u[1] = bp[1];
      acc = __builtin_amdgcn_wmma_f32_16x16x32_bf16(
                false, a.v, false, bm.v, (short)0, acc, false, false);
    }
    // D layout: lane holds column e = et*16 + (lane&15); VGPR g = row g (+8 hi half)
    const int   e  = et * 16 + (lane & 15);
    const float bb = biasS[e];
    const float ww = w2S[e];
    #pragma unroll
    for (int g = 0; g < 8; ++g) sc[g] += fast_tanh(acc[g] + bb) * ww;
  }

  // reduce over the 16 lanes that share each row (stay within wave halves)
  #pragma unroll
  for (int g = 0; g < 8; ++g) {
    float v = sc[g];
    v += __shfl_xor(v, 1);
    v += __shfl_xor(v, 2);
    v += __shfl_xor(v, 4);
    v += __shfl_xor(v, 8);
    sc[g] = v;
  }
  if ((lane & 15) == 0) {
    const int base = s0w + hi * 8;
    #pragma unroll
    for (int g = 0; g < 8; ++g) scores[b * S_ + base + g] = sc[g];
  }
}

// ---------------------------------------------------------------------------
// Kernel 4: softmax over S per batch (one block per batch, values in registers)
// ---------------------------------------------------------------------------
__global__ void softmax_kernel(const float* __restrict__ scores,
                               float* __restrict__ attn) {
  __shared__ float red[256];
  const int b = blockIdx.x, t = threadIdx.x;
  float v[16];
  float mx = -1e30f;
  #pragma unroll
  for (int i = 0; i < 16; ++i) {
    v[i] = scores[b * S_ + t + i * 256];
    mx = fmaxf(mx, v[i]);
  }
  red[t] = mx; __syncthreads();
  for (int off = 128; off > 0; off >>= 1) {
    if (t < off) red[t] = fmaxf(red[t], red[t + off]);
    __syncthreads();
  }
  mx = red[0]; __syncthreads();
  float sum = 0.f;
  #pragma unroll
  for (int i = 0; i < 16; ++i) { v[i] = __expf(v[i] - mx); sum += v[i]; }
  red[t] = sum; __syncthreads();
  for (int off = 128; off > 0; off >>= 1) {
    if (t < off) red[t] += red[t + off];
    __syncthreads();
  }
  const float inv = 1.0f / red[0];
  #pragma unroll
  for (int i = 0; i < 16; ++i) attn[b * S_ + t + i * 256] = v[i] * inv;
}

// ---------------------------------------------------------------------------
// Kernel 5: mix[b,e] = sum_s attn[b,s] * context[b,s,e]   (memory bound)
// grid = B * 4 (128 e-columns per block); context read non-temporally.
// ---------------------------------------------------------------------------
__global__ void mix_kernel(const float* __restrict__ attn,
                           const float* __restrict__ ctx_g,
                           float* __restrict__ mix) {
  __shared__ float red[256];
  const int b  = blockIdx.x >> 2;
  const int e0 = (blockIdx.x & 3) * 128;
  const int t  = threadIdx.x;
  const int e  = e0 + (t & 127);
  const int s0 = t >> 7;
  const float* cp = ctx_g + (size_t)b * S_ * HE_ + e;
  const float* ap = attn + b * S_;
  float acc = 0.f;
  for (int s = s0; s < S_; s += 2)
    acc += ap[s] * __builtin_nontemporal_load(&cp[(size_t)s * HE_]);
  red[t] = acc; __syncthreads();
  if (t < 128) mix[b * HE_ + e0 + t] = red[t] + red[t + 128];
}

// ---------------------------------------------------------------------------
// Kernel 6: out[b,:] = tanh(concat(mix, output) @ Wout^T + bout)
// ---------------------------------------------------------------------------
__global__ void final_kernel(const float* __restrict__ mix,
                             const float* __restrict__ outp,
                             const float* __restrict__ Wout,
                             const float* __restrict__ bout,
                             float* __restrict__ out) {
  __shared__ float comb[HD_];
  const int b = blockIdx.x, t = threadIdx.x;   // 256 threads
  comb[t]         = mix[b * HE_ + t];
  comb[t + 256]   = mix[b * HE_ + t + 256];
  comb[512 + t]       = outp[b * 512 + t];
  comb[512 + t + 256] = outp[b * 512 + t + 256];
  __syncthreads();
  for (int e = t; e < HE_; e += 256) {
    const float* w = Wout + (size_t)e * HD_;
    float acc = bout[e];
    for (int d = 0; d < HD_; ++d) acc += comb[d] * w[d];
    out[b * HE_ + e] = fast_tanh(acc);
  }
}

// ---------------------------------------------------------------------------
extern "C" void kernel_launch(void* const* d_in, const int* in_sizes, int n_in,
                              void* d_out, int out_size, void* d_ws, size_t ws_size,
                              hipStream_t stream) {
  const float* outp = (const float*)d_in[0];  // [B,1,DOUT]
  const float* ctx  = (const float*)d_in[1];  // [B,S,HE]
  const float* W1   = (const float*)d_in[2];  // [HE,HD]
  const float* W2   = (const float*)d_in[3];  // [1,HE]
  const float* Wout = (const float*)d_in[4];  // [HE,HD]
  const float* bout = (const float*)d_in[5];  // [HE]
  float* out = (float*)d_out;                 // [B,1,HE]

  char* ws = (char*)d_ws;
  unsigned int* W1b = (unsigned int*)(ws);                        // 512 KB
  float* bias   = (float*)(ws + (size_t)(512) * 1024);            //  64 KB
  float* scores = (float*)(ws + (size_t)(576) * 1024);            // 512 KB
  float* attn   = (float*)(ws + (size_t)(1088) * 1024);           // 512 KB
  float* mix    = (float*)(ws + (size_t)(1600) * 1024);           //  64 KB

  pack_w1_kernel<<<131072 / 256, 256, 0, stream>>>(W1, W1b);
  bias_kernel<<<B_, 256, 0, stream>>>(outp, W1, bias);
  score_kernel<<<B_ * (S_ / 128), 256, SCORE_SMEM, stream>>>(ctx, W1b, bias, W2, scores);
  softmax_kernel<<<B_, 256, 0, stream>>>(scores, attn);
  mix_kernel<<<B_ * 4, 256, 0, stream>>>(attn, ctx, mix);
  final_kernel<<<B_, 256, 0, stream>>>(mix, outp, Wout, bout, out);
}
